// RealMPNNLayer_292057776274
// MI455X (gfx1250) — matrix operand order repaired and checked by reading, not compiled, not run on hardware
//
#include <hip/hip_runtime.h>

#define N_NODES 50000
#define N_EDGES 800000
#define ND 128
#define ED 64
#define HD 128

typedef __attribute__((ext_vector_type(16))) __bf16 v16bf;
typedef __attribute__((ext_vector_type(8)))  __bf16 v8bf;
typedef __attribute__((ext_vector_type(8)))  float  v8f;

// ---------------- helpers ----------------

__device__ __forceinline__ v8f splat8(float v) {
  v8f r; r[0]=v;r[1]=v;r[2]=v;r[3]=v;r[4]=v;r[5]=v;r[6]=v;r[7]=v; return r;
}

// Build a 16-element bf16 A-fragment from two 8-float contiguous chunks.
__device__ __forceinline__ v16bf cvt_chunks(const float* __restrict__ p0,
                                            const float* __restrict__ p1) {
  float4 a0 = *(const float4*)(p0);
  float4 a1 = *(const float4*)(p0 + 4);
  float4 b0 = *(const float4*)(p1);
  float4 b1 = *(const float4*)(p1 + 4);
  v16bf r;
  r[0]=(__bf16)a0.x; r[1]=(__bf16)a0.y; r[2]=(__bf16)a0.z; r[3]=(__bf16)a0.w;
  r[4]=(__bf16)a1.x; r[5]=(__bf16)a1.y; r[6]=(__bf16)a1.z; r[7]=(__bf16)a1.w;
  r[8]=(__bf16)b0.x; r[9]=(__bf16)b0.y; r[10]=(__bf16)b0.z; r[11]=(__bf16)b0.w;
  r[12]=(__bf16)b1.x; r[13]=(__bf16)b1.y; r[14]=(__bf16)b1.z; r[15]=(__bf16)b1.w;
  return r;
}

// Source pointer inside concat(node[src], node[dst], edge) for an 8-aligned chunk.
__device__ __forceinline__ const float* msg_ptr(int kk,
    const float* __restrict__ node, const float* __restrict__ edgef,
    int s, int d, long e) {
  if (kk < ND)        return node  + (long)s * ND + kk;
  else if (kk < 2*ND) return node  + (long)d * ND + (kk - ND);
  else                return edgef + e * ED + (kk - 2*ND);
}

// ---------------- prep: zero agg, pack weights to bf16 B-fragment layout ----------------
// Fragment layout (per 32x16 K/N tile): lane = 16*((k%16)/8) + (n%16),
// element i = (k%8) + 8*((k%32)/16), fragment index = (k/32)*8 + n/16, 512 bf16 per fragment.

__device__ __forceinline__ void packW(const float* __restrict__ W, __bf16* __restrict__ o,
                                      int K, long tid, long stride) {
  const long total = (long)K * 128;
  for (long idx = tid; idx < total; idx += stride) {
    int k  = (int)(idx >> 7);
    int n  = (int)(idx & 127);
    int kt = k >> 5, kr = k & 31, hh = (kr >> 3) & 1, i = (kr & 7) | ((kr >> 4) << 3);
    int t  = n >> 4, nn2 = n & 15;
    o[(((long)kt * 8 + t) << 9) + (hh * 16 + nn2) * 16 + i] = (__bf16)W[idx];
  }
}

__global__ void mpnn_prep_kernel(const float* __restrict__ Wm1, const float* __restrict__ Wm2,
                                 const float* __restrict__ Wu1, const float* __restrict__ Wu2,
                                 float* __restrict__ agg,
                                 __bf16* __restrict__ wm1b, __bf16* __restrict__ wm2b,
                                 __bf16* __restrict__ wu1b, __bf16* __restrict__ wu2b) {
  long tid    = (long)blockIdx.x * blockDim.x + threadIdx.x;
  long stride = (long)gridDim.x * blockDim.x;
  for (long i = tid; i < (long)N_NODES * HD; i += stride) agg[i] = 0.0f;
  packW(Wm1, wm1b, 2*ND + ED, tid, stride);
  packW(Wm2, wm2b, HD,        tid, stride);
  packW(Wu1, wu1b, ND + HD,   tid, stride);
  packW(Wu2, wu2b, HD,        tid, stride);
}

// ---------------- kernel 1: edge message MLP + scatter-add ----------------
// Wave handles M=32 edges (two 16-row tiles sharing each B fragment) x N=128.

__global__ __launch_bounds__(256) void mpnn_msg_kernel(
    const float* __restrict__ node, const int* __restrict__ srcI,
    const int* __restrict__ dstI, const float* __restrict__ edgef,
    const __bf16* __restrict__ wm1b, const float* __restrict__ bm1,
    const __bf16* __restrict__ wm2b, const float* __restrict__ bm2,
    float* __restrict__ agg) {
  __shared__ __align__(16) __bf16 hbuf[8 * 32 * HD];   // 64 KB: per-wave 32x128 staging
  const int lane = threadIdx.x & 31;
  const int wave = threadIdx.x >> 5;
  const int nn   = lane & 15;
  const int h    = lane >> 4;
  const long eb  = (long)blockIdx.x * 256 + wave * 32;
  const long e0  = eb + nn;
  const long e1  = eb + 16 + nn;
  const int s0   = srcI[e0];
  const int d0   = dstI[e0];
  const int s1   = srcI[e1];
  const int d1   = dstI[e1];

  // ---- GEMM 1: [32 x 320] x [320 x 128] ----
  v8f acc[2][8];
  #pragma unroll
  for (int t = 0; t < 8; t++) {
    const float bv = bm1[t * 16 + nn];
    acc[0][t] = splat8(bv);
    acc[1][t] = splat8(bv);
  }

  #pragma unroll
  for (int kt = 0; kt < 10; kt++) {
    const int k0 = kt * 32 + 8 * h;
    const int k1 = k0 + 16;
    v16bf a0 = cvt_chunks(msg_ptr(k0, node, edgef, s0, d0, e0),
                          msg_ptr(k1, node, edgef, s0, d0, e0));
    v16bf a1 = cvt_chunks(msg_ptr(k0, node, edgef, s1, d1, e1),
                          msg_ptr(k1, node, edgef, s1, d1, e1));
    #pragma unroll
    for (int t = 0; t < 8; t++) {
      v16bf bfrag = *(const v16bf*)(wm1b + (((long)kt * 8 + t) << 9) + lane * 16);
      acc[0][t] = __builtin_amdgcn_wmma_f32_16x16x32_bf16(
          false, a0, false, bfrag, (short)0, acc[0][t], false, false);
      acc[1][t] = __builtin_amdgcn_wmma_f32_16x16x32_bf16(
          false, a1, false, bfrag, (short)0, acc[1][t], false, false);
    }
  }

  // ReLU, stage to per-wave LDS in row-major bf16 for A-fragment reads
  __bf16* hb = hbuf + wave * (32 * HD);
  #pragma unroll
  for (int mi = 0; mi < 2; mi++) {
    #pragma unroll
    for (int t = 0; t < 8; t++) {
      #pragma unroll
      for (int r = 0; r < 8; r++) {
        hb[(mi * 16 + r + 8 * h) * HD + t * 16 + nn] = (__bf16)fmaxf(acc[mi][t][r], 0.0f);
      }
    }
  }

  // ---- GEMM 2: [32 x 128] x [128 x 128] ----
  v8f acc2[2][8];
  #pragma unroll
  for (int t = 0; t < 8; t++) {
    const float bv = bm2[t * 16 + nn];
    acc2[0][t] = splat8(bv);
    acc2[1][t] = splat8(bv);
  }

  #pragma unroll
  for (int kt = 0; kt < 4; kt++) {
    const int k0 = kt * 32 + 8 * h;
    v16bf a[2];
    #pragma unroll
    for (int mi = 0; mi < 2; mi++) {
      const __bf16* hrow = hb + (mi * 16 + nn) * HD;
      v8bf c0 = *(const v8bf*)(hrow + k0);
      v8bf c1 = *(const v8bf*)(hrow + k0 + 16);
      a[mi] = __builtin_shufflevector(c0, c1,
          0,1,2,3,4,5,6,7,8,9,10,11,12,13,14,15);
    }
    #pragma unroll
    for (int t = 0; t < 8; t++) {
      v16bf bfrag = *(const v16bf*)(wm2b + (((long)kt * 8 + t) << 9) + lane * 16);
      acc2[0][t] = __builtin_amdgcn_wmma_f32_16x16x32_bf16(
          false, a[0], false, bfrag, (short)0, acc2[0][t], false, false);
      acc2[1][t] = __builtin_amdgcn_wmma_f32_16x16x32_bf16(
          false, a[1], false, bfrag, (short)0, acc2[1][t], false, false);
    }
  }

  // ---- scatter-add messages to destination nodes ----
  #pragma unroll
  for (int mi = 0; mi < 2; mi++) {
    const int dmi = mi ? d1 : d0;
    #pragma unroll
    for (int r = 0; r < 8; r++) {
      const int row  = r + 8 * h;
      const int drow = __shfl(dmi, row, 32);
      float* ap = agg + (long)drow * HD + nn;
      #pragma unroll
      for (int t = 0; t < 8; t++) {
        atomicAdd(ap + t * 16, acc2[mi][t][r]);
      }
    }
  }
}

// ---------------- kernel 2: node update MLP + residual + LayerNorm ----------------

__global__ __launch_bounds__(256) void mpnn_upd_kernel(
    const float* __restrict__ node, const float* __restrict__ agg,
    const __bf16* __restrict__ wu1b, const float* __restrict__ bu1,
    const __bf16* __restrict__ wu2b, const float* __restrict__ bu2,
    const float* __restrict__ gamma, const float* __restrict__ beta,
    float* __restrict__ out) {
  __shared__ __align__(16) __bf16 hbuf[8 * 32 * HD];
  const int lane = threadIdx.x & 31;
  const int wave = threadIdx.x >> 5;
  const int nn   = lane & 15;
  const int h    = lane >> 4;
  const long nb  = (long)blockIdx.x * 256 + wave * 32;
  long vcl[2];
  #pragma unroll
  for (int mi = 0; mi < 2; mi++) {
    const long v = nb + mi * 16 + nn;
    vcl[mi] = (v < N_NODES) ? v : (long)(N_NODES - 1);
  }

  // ---- GEMM 1: [32 x 256] x [256 x 128] (A = concat(node, agg)) ----
  v8f acc[2][8];
  #pragma unroll
  for (int t = 0; t < 8; t++) {
    const float bv = bu1[t * 16 + nn];
    acc[0][t] = splat8(bv);
    acc[1][t] = splat8(bv);
  }

  #pragma unroll
  for (int kt = 0; kt < 8; kt++) {
    const int k0 = kt * 32 + 8 * h;
    const int k1 = k0 + 16;
    v16bf a[2];
    #pragma unroll
    for (int mi = 0; mi < 2; mi++) {
      const float* p0 = (k0 < ND) ? (node + vcl[mi] * ND + k0)
                                  : (agg + vcl[mi] * HD + (k0 - ND));
      const float* p1 = (k1 < ND) ? (node + vcl[mi] * ND + k1)
                                  : (agg + vcl[mi] * HD + (k1 - ND));
      a[mi] = cvt_chunks(p0, p1);
    }
    #pragma unroll
    for (int t = 0; t < 8; t++) {
      v16bf bfrag = *(const v16bf*)(wu1b + (((long)kt * 8 + t) << 9) + lane * 16);
      acc[0][t] = __builtin_amdgcn_wmma_f32_16x16x32_bf16(
          false, a[0], false, bfrag, (short)0, acc[0][t], false, false);
      acc[1][t] = __builtin_amdgcn_wmma_f32_16x16x32_bf16(
          false, a[1], false, bfrag, (short)0, acc[1][t], false, false);
    }
  }

  __bf16* hb = hbuf + wave * (32 * HD);
  #pragma unroll
  for (int mi = 0; mi < 2; mi++) {
    #pragma unroll
    for (int t = 0; t < 8; t++) {
      #pragma unroll
      for (int r = 0; r < 8; r++) {
        hb[(mi * 16 + r + 8 * h) * HD + t * 16 + nn] = (__bf16)fmaxf(acc[mi][t][r], 0.0f);
      }
    }
  }

  // ---- GEMM 2: [32 x 128] x [128 x 128] ----
  v8f acc2[2][8];
  #pragma unroll
  for (int t = 0; t < 8; t++) {
    const float bv = bu2[t * 16 + nn];
    acc2[0][t] = splat8(bv);
    acc2[1][t] = splat8(bv);
  }

  #pragma unroll
  for (int kt = 0; kt < 4; kt++) {
    const int k0 = kt * 32 + 8 * h;
    v16bf a[2];
    #pragma unroll
    for (int mi = 0; mi < 2; mi++) {
      const __bf16* hrow = hb + (mi * 16 + nn) * HD;
      v8bf c0 = *(const v8bf*)(hrow + k0);
      v8bf c1 = *(const v8bf*)(hrow + k0 + 16);
      a[mi] = __builtin_shufflevector(c0, c1,
          0,1,2,3,4,5,6,7,8,9,10,11,12,13,14,15);
    }
    #pragma unroll
    for (int t = 0; t < 8; t++) {
      v16bf bfrag = *(const v16bf*)(wu2b + (((long)kt * 8 + t) << 9) + lane * 16);
      acc2[0][t] = __builtin_amdgcn_wmma_f32_16x16x32_bf16(
          false, a[0], false, bfrag, (short)0, acc2[0][t], false, false);
      acc2[1][t] = __builtin_amdgcn_wmma_f32_16x16x32_bf16(
          false, a[1], false, bfrag, (short)0, acc2[1][t], false, false);
    }
  }

  // ---- residual + LayerNorm (row mi*16 + r + 8h owned by the 16-lane half h) ----
  float gv[8], bv[8];
  #pragma unroll
  for (int t = 0; t < 8; t++) { gv[t] = gamma[t * 16 + nn]; bv[t] = beta[t * 16 + nn]; }

  #pragma unroll
  for (int mi = 0; mi < 2; mi++) {
    #pragma unroll
    for (int r = 0; r < 8; r++) {
      const int row = mi * 16 + r + 8 * h;
      const long vr = nb + row;
      const long vc = (vr < N_NODES) ? vr : (long)(N_NODES - 1);
      const float* resp = node + vc * ND + nn;
      float x[8];
      float s1 = 0.0f, s2 = 0.0f;
      #pragma unroll
      for (int t = 0; t < 8; t++) {
        x[t] = acc2[mi][t][r] + resp[t * 16];
        s1 += x[t];
        s2 += x[t] * x[t];
      }
      #pragma unroll
      for (int msk = 1; msk < 16; msk <<= 1) {
        s1 += __shfl_xor(s1, msk, 32);
        s2 += __shfl_xor(s2, msk, 32);
      }
      const float mean = s1 * (1.0f / 128.0f);
      const float var  = s2 * (1.0f / 128.0f) - mean * mean;
      const float rstd = rsqrtf(var + 1e-5f);
      if (vr < N_NODES) {
        float* op = out + vr * ND + nn;
        #pragma unroll
        for (int t = 0; t < 8; t++) {
          op[t * 16] = (x[t] - mean) * rstd * gv[t] + bv[t];
        }
      }
    }
  }
}

// ---------------- launch ----------------

extern "C" void kernel_launch(void* const* d_in, const int* in_sizes, int n_in,
                              void* d_out, int out_size, void* d_ws, size_t ws_size,
                              hipStream_t stream) {
  (void)in_sizes; (void)n_in; (void)out_size; (void)ws_size;
  const float* node  = (const float*)d_in[0];
  const int*   eidx  = (const int*)d_in[1];
  const float* edgef = (const float*)d_in[2];
  const float* Wm1   = (const float*)d_in[3];
  const float* bm1   = (const float*)d_in[4];
  const float* Wm2   = (const float*)d_in[5];
  const float* bm2   = (const float*)d_in[6];
  const float* Wu1   = (const float*)d_in[7];
  const float* bu1   = (const float*)d_in[8];
  const float* Wu2   = (const float*)d_in[9];
  const float* bu2   = (const float*)d_in[10];
  const float* gamma = (const float*)d_in[11];
  const float* beta  = (const float*)d_in[12];
  float* out = (float*)d_out;

  const int* src = eidx;            // edge_index row 0
  const int* dst = eidx + N_EDGES;  // edge_index row 1

  // workspace layout (all offsets 32B aligned)
  char* ws = (char*)d_ws;
  float*  agg  = (float*)ws;                            // 50000*128 f32 = 25,600,000 B
  __bf16* wm1b = (__bf16*)(ws + 25600000);              // 320*128 bf16 =     81,920 B
  __bf16* wm2b = (__bf16*)(ws + 25681920);              // 128*128 bf16 =     32,768 B
  __bf16* wu1b = (__bf16*)(ws + 25714688);              // 256*128 bf16 =     65,536 B
  __bf16* wu2b = (__bf16*)(ws + 25780224);              // 128*128 bf16 =     32,768 B

  mpnn_prep_kernel<<<512, 256, 0, stream>>>(Wm1, Wm2, Wu1, Wu2,
                                            agg, wm1b, wm2b, wu1b, wu2b);

  mpnn_msg_kernel<<<N_EDGES / 256, 256, 0, stream>>>(
      node, src, dst, edgef, wm1b, bm1, wm2b, bm2, agg);

  mpnn_upd_kernel<<<(N_NODES + 255) / 256, 256, 0, stream>>>(
      node, agg, wu1b, bu1, wu2b, bu2, gamma, beta, out);
}
